// Decoder3D_20581483282611
// MI455X (gfx1250) — matrix operand order repaired and checked by reading, not compiled, not run on hardware
//
#include <hip/hip_runtime.h>
#include <hip/hip_bf16.h>

typedef __attribute__((ext_vector_type(16))) _Float16 v16h;
typedef __attribute__((ext_vector_type(8)))  float    v8f;
typedef unsigned int u32;
typedef __attribute__((ext_vector_type(4))) unsigned int u32x4;
typedef __attribute__((ext_vector_type(8))) int         i32x8;
typedef __attribute__((ext_vector_type(4))) int         i32x4;

#define WPB 4          // waves per block (wave32)
#define TPB (WPB * 32) // 128 threads
#define KC  64         // K-chunk staged in LDS (2 WMMA steps)

#if __has_builtin(__builtin_amdgcn_tensor_load_to_lds) && \
    __has_builtin(__builtin_amdgcn_s_wait_tensorcnt)
#define HAVE_TDM 1
#else
#define HAVE_TDM 0
#endif

union HFrag { v16h v; _Float16 e[16]; };

// A-fragment (16x32 f16, M x K) per ISA VGPR layout.
__device__ __forceinline__ v16h lds_afrag(const _Float16* As, int k0) {
  const int lane = threadIdx.x & 31;
  const int row  = lane & 15;
  const int hi   = lane >> 4;
  const _Float16* p = As + row * KC + k0 + hi * 8;
  HFrag f;
#pragma unroll
  for (int i = 0; i < 8; ++i) { f.e[i] = p[i]; f.e[8 + i] = p[16 + i]; }
  return f.v;
}

// B-fragment (32x16 f16, K x N); Bs stored as [N=16][KC].
__device__ __forceinline__ v16h lds_bfrag(const _Float16* Bs, int k0) {
  const int lane = threadIdx.x & 31;
  const int col  = lane & 15;
  const int hi   = lane >> 4;
  const _Float16* p = Bs + col * KC + k0 + hi * 8;
  HFrag f;
#pragma unroll
  for (int i = 0; i < 8; ++i) { f.e[i] = p[i]; f.e[8 + i] = p[16 + i]; }
  return f.v;
}

__device__ __forceinline__ v8f wmma_chunk(const _Float16* As, const _Float16* Bw, v8f acc) {
#pragma unroll
  for (int ks = 0; ks < KC; ks += 32) {
    v16h a = lds_afrag(As, ks);
    v16h b = lds_bfrag(Bw, ks);
    acc = __builtin_amdgcn_wmma_f32_16x16x32_f16(false, a, false, b, (short)0, acc,
                                                 false, false);
  }
  return acc;
}

#if HAVE_TDM
// Issue one 2D TDM tile load: tileH(<=16) x tileW(f16 elems) from a row-major
// [*, strideElems] f16 matrix into LDS at lds_off. remW/remH are the remaining
// tensor extent measured from the tile origin so TDM's OOB logic zero-fills the
// K tail and row padding.
__device__ __forceinline__ void tdm_load_2d(u32 lds_off, const void* gptr, u32 remW,
                                            u32 remH, u32 tileW, u32 tileH,
                                            u32 strideElems) {
  unsigned long long ga = (unsigned long long)(uintptr_t)gptr;
  u32x4 g0;
  g0[0] = 1u;                                        // count=1 (valid user D#)
  g0[1] = lds_off;                                   // lds_addr (bytes)
  g0[2] = (u32)(ga & 0xFFFFFFFFull);                 // global_addr[31:0]
  g0[3] = (u32)((ga >> 32) & 0x1FFFFFFull) | (2u << 30); // addr[56:32] | type=2
  i32x8 g1;
  g1[0] = (int)(1u << 16);                           // data_size=1 (2 bytes/elem)
  g1[1] = (int)((remW & 0xFFFFu) << 16);             // tensor_dim0[15:0]
  g1[2] = (int)(((remW >> 16) & 0xFFFFu) | ((remH & 0xFFFFu) << 16));
  g1[3] = (int)(((remH >> 16) & 0xFFFFu) | ((tileW & 0xFFFFu) << 16)); // tile_dim0
  g1[4] = (int)(tileH & 0xFFFFu);                    // tile_dim1 (tile_dim2=0)
  g1[5] = (int)strideElems;                          // tensor_dim0_stride[31:0]
  g1[6] = 0;                                         // stride[47:32], dim1_stride
  g1[7] = 0;
  i32x4 z4 = {0, 0, 0, 0};                           // groups 2/3 unused (2D)
  i32x8 z8 = {0, 0, 0, 0, 0, 0, 0, 0};
  __builtin_amdgcn_tensor_load_to_lds(g0, g1, z4, z4, z8, 0);
}
__device__ __forceinline__ u32 lds_off_of(const void* p) {
  // generic LDS address: low 32 bits are the wave-relative LDS byte offset
  return (u32)(uintptr_t)p;
}
#endif

// ---------------- f32 -> f16 weight conversion ----------------
__global__ void k_f32_to_f16(_Float16* __restrict__ dst, const float* __restrict__ src,
                             int n) {
  int i = blockIdx.x * blockDim.x + threadIdx.x;
  if (i < n) dst[i] = (_Float16)src[i];
}

// ---------------- fused grid_sample (1x1 conv commuted past it) ----------------
__global__ void k_sample(const float* __restrict__ kp, const float* __restrict__ x,
                         _Float16* __restrict__ xs) {
  const int bk = blockIdx.x;            // 0..1343
  const int b = bk / 21, k = bk - b * 21;
  const int c = threadIdx.x;            // 0..255
  const float u = kp[(b * 21 + k) * 2 + 0];
  const float w = kp[(b * 21 + k) * 2 + 1];
  const float gx = fminf(fmaxf((u - 0.5f) * 2.0f, -1.0f), 1.0f);
  const float gy = fminf(fmaxf((w - 0.5f) * 2.0f, -1.0f), 1.0f);
  const float px = (gx + 1.0f) * 0.5f * 63.0f;
  const float py = (gy + 1.0f) * 0.5f * 63.0f;
  const float x0f = floorf(px), y0f = floorf(py);
  const float wx = px - x0f, wy = py - y0f;
  const int x0 = (int)x0f, y0 = (int)y0f;
  const float* base = x + ((size_t)b * 256 + c) * 4096;
  auto g = [&](int yi, int xi) -> float {  // branch-free, always-valid load
    int lin = (yi < 63 ? yi : 63) * 64 + (xi < 63 ? xi : 63);
    float v = base[lin < 0 ? 0 : lin];
    bool ok = (xi >= 0) & (xi <= 63) & (yi >= 0) & (yi <= 63);
    return ok ? v : 0.0f;
  };
  const float v00 = g(y0, x0), v01 = g(y0, x0 + 1);
  const float v10 = g(y0 + 1, x0), v11 = g(y0 + 1, x0 + 1);
  const float r = (v00 * (1.0f - wx) + v01 * wx) * (1.0f - wy) +
                  (v10 * (1.0f - wx) + v11 * wx) * wy;
  xs[(size_t)bk * 256 + c] = (_Float16)r;
}

// ---------------- plain GEMM: out[M,N] = A[M,K] * W[N,K]^T + bias ----------------
__global__ __launch_bounds__(TPB) void k_gemm(const _Float16* __restrict__ A,
                                              const _Float16* __restrict__ Wg,
                                              const float* __restrict__ bias,
                                              _Float16* __restrict__ out, int M,
                                              int Ktot, int Nout) {
  __shared__ _Float16 As[16 * KC];
  __shared__ _Float16 Bs[WPB * 16 * KC];
  const int tid = threadIdx.x, wave = tid >> 5;
  const int mbase = blockIdx.x * 16;
  const int nbase = (blockIdx.y * WPB + wave) * 16;
  v8f acc = {};
  for (int k0 = 0; k0 < Ktot; k0 += KC) {
#if HAVE_TDM
    {
      int rh = Nout - nbase;
      u32 remH = rh > 0 ? (u32)rh : 0u;
      const _Float16* src = (rh > 0) ? (Wg + (size_t)nbase * Ktot + k0) : Wg;
      tdm_load_2d(lds_off_of(Bs + wave * 16 * KC), src, (u32)(Ktot - k0), remH, KC, 16,
                  (u32)Ktot);
    }
#endif
    for (int t = tid; t < 16 * KC; t += TPB) {  // branch-free A fill
      int r = t >> 6, kk = t & (KC - 1);
      int row = mbase + r, k = k0 + kk;
      bool ok = (row < M) & (k < Ktot);
      int rc = row < M ? row : M - 1;
      int kc = k < Ktot ? k : Ktot - 1;
      _Float16 val = A[(size_t)rc * Ktot + kc];
      As[t] = ok ? val : (_Float16)0.0f;
    }
#if !HAVE_TDM
    for (int t = tid; t < WPB * 16 * KC; t += TPB) {
      int n = blockIdx.y * WPB * 16 + (t >> 6);
      int k = k0 + (t & (KC - 1));
      bool ok = (n < Nout) & (k < Ktot);
      int nc = n < Nout ? n : Nout - 1;
      int kc = k < Ktot ? k : Ktot - 1;
      _Float16 val = Wg[(size_t)nc * Ktot + kc];
      Bs[t] = ok ? val : (_Float16)0.0f;
    }
#else
    __builtin_amdgcn_s_wait_tensorcnt(0);
#endif
    __syncthreads();
    acc = wmma_chunk(As, Bs + wave * 16 * KC, acc);
    __syncthreads();
  }
  const int lane = tid & 31;
  const int N = nbase + (lane & 15);
  if (N >= Nout) return;
  const float bv = bias ? bias[N] : 0.0f;
  const int mo = (lane >> 4) << 3;
#pragma unroll
  for (int v = 0; v < 8; ++v) {
    int Mr = mbase + mo + v;
    if (Mr < M) out[(size_t)Mr * Nout + N] = (_Float16)(acc[v] + bv);
  }
}

// ---------------- mesh upsample: hout[b,v,c] = sum_u ut[v,u] * hin[b,u,c] ----------------
__global__ __launch_bounds__(TPB) void k_upsample(const _Float16* __restrict__ ut,
                                                  const _Float16* __restrict__ hin,
                                                  _Float16* __restrict__ hout, int Vout,
                                                  int Vin, int C) {
  __shared__ _Float16 As[16 * KC];
  __shared__ _Float16 Bs[WPB * 16 * KC];
  const int tid = threadIdx.x, wave = tid >> 5;
  const int mtiles = (Vout + 15) >> 4;
  const int b = blockIdx.x / mtiles;
  const int vbase = (blockIdx.x - b * mtiles) * 16;
  const int nbase = (blockIdx.y * WPB + wave) * 16;
  const _Float16* hb = hin + (size_t)b * Vin * C;
  const int Kloop = ((Vin + KC - 1) / KC) * KC;
  v8f acc = {};
  for (int k0 = 0; k0 < Kloop; k0 += KC) {
    for (int t = tid; t < 16 * KC; t += TPB) {  // branch-free A fill
      int r = t >> 6, kk = t & (KC - 1);
      int v = vbase + r, u = k0 + kk;
      bool ok = (v < Vout) & (u < Vin);
      int vc = v < Vout ? v : Vout - 1;
      int uc = u < Vin ? u : Vin - 1;
      _Float16 val = ut[(size_t)vc * Vin + uc];
      As[t] = ok ? val : (_Float16)0.0f;
    }
    for (int t = tid; t < WPB * 16 * KC; t += TPB) {  // transposed B gather (no TDM)
      int c = blockIdx.y * WPB * 16 + (t >> 6);
      int u = k0 + (t & (KC - 1));
      bool ok = (c < C) & (u < Vin);
      int cc = c < C ? c : C - 1;
      int uc = u < Vin ? u : Vin - 1;
      _Float16 val = hb[(size_t)uc * C + cc];
      Bs[t] = ok ? val : (_Float16)0.0f;
    }
    __syncthreads();
    acc = wmma_chunk(As, Bs + wave * 16 * KC, acc);
    __syncthreads();
  }
  const int lane = tid & 31;
  const int N = nbase + (lane & 15);
  if (N >= C) return;
  _Float16* ob = hout + (size_t)b * Vout * C;
  const int mo = (lane >> 4) << 3;
#pragma unroll
  for (int v = 0; v < 8; ++v) {
    int Mr = vbase + mo + v;
    if (Mr < Vout) ob[(size_t)Mr * C + N] = (_Float16)acc[v];
  }
}

// ---------------- spiral conv: gather-in-LDS + GEMM + bias (+relu) ----------------
__global__ __launch_bounds__(TPB) void k_dsconv(const _Float16* __restrict__ hin,
                                                const int* __restrict__ idx,
                                                const _Float16* __restrict__ Wg,
                                                const float* __restrict__ bias,
                                                _Float16* __restrict__ outh,
                                                float* __restrict__ outf, int V,
                                                int cshift, int Nout, int relu) {
  __shared__ _Float16 As[16 * KC];
  __shared__ _Float16 Bs[WPB * 16 * KC];
  const int Cin = 1 << cshift;
  const int Ktot = 9 << cshift; // S = 9
  const int tid = threadIdx.x, wave = tid >> 5;
  const int mtiles = (V + 15) >> 4;
  const int b = blockIdx.x / mtiles;
  const int vbase = (blockIdx.x - b * mtiles) * 16;
  const int nbase = (blockIdx.y * WPB + wave) * 16;
  const _Float16* hb = hin + (size_t)b * V * Cin;
  v8f acc = {};
  for (int k0 = 0; k0 < Ktot; k0 += KC) {
#if HAVE_TDM
    {
      int rh = Nout - nbase;
      u32 remH = rh > 0 ? (u32)rh : 0u;
      const _Float16* src = (rh > 0) ? (Wg + (size_t)nbase * Ktot + k0) : Wg;
      tdm_load_2d(lds_off_of(Bs + wave * 16 * KC), src, (u32)(Ktot - k0), remH, KC, 16,
                  (u32)Ktot);
    }
#endif
    for (int t = tid; t < 16 * KC; t += TPB) {  // branch-free spiral gather
      int r = t >> 6, kk = t & (KC - 1);
      int v = vbase + r, k = k0 + kk;
      bool ok = (v < V) & (k < Ktot);
      int vc = v < V ? v : V - 1;
      int kc = k < Ktot ? k : 0;
      int s = kc >> cshift, c = kc & (Cin - 1);
      _Float16 val = hb[(size_t)idx[vc * 9 + s] * Cin + c];
      As[t] = ok ? val : (_Float16)0.0f;
    }
#if !HAVE_TDM
    for (int t = tid; t < WPB * 16 * KC; t += TPB) {
      int n = blockIdx.y * WPB * 16 + (t >> 6);
      int k = k0 + (t & (KC - 1));
      bool ok = (n < Nout) & (k < Ktot);
      int nc = n < Nout ? n : Nout - 1;
      int kc = k < Ktot ? k : Ktot - 1;
      _Float16 val = Wg[(size_t)nc * Ktot + kc];
      Bs[t] = ok ? val : (_Float16)0.0f;
    }
#else
    __builtin_amdgcn_s_wait_tensorcnt(0);
#endif
    __syncthreads();
    acc = wmma_chunk(As, Bs + wave * 16 * KC, acc);
    __syncthreads();
  }
  const int lane = tid & 31;
  const int N = nbase + (lane & 15);
  if (N >= Nout) return;
  const float bv = bias[N];
  const int mo = (lane >> 4) << 3;
  if (outf) {
    float* ob = outf + (size_t)b * V * Nout;
#pragma unroll
    for (int v = 0; v < 8; ++v) {
      int Mr = vbase + mo + v;
      float val = acc[v] + bv;
      if (relu) val = fmaxf(val, 0.0f);
      if (Mr < V) ob[(size_t)Mr * Nout + N] = val;
    }
  } else {
    _Float16* ob = outh + (size_t)b * V * Nout;
#pragma unroll
    for (int v = 0; v < 8; ++v) {
      int Mr = vbase + mo + v;
      float val = acc[v] + bv;
      if (relu) val = fmaxf(val, 0.0f);
      if (Mr < V) ob[(size_t)Mr * Nout + N] = (_Float16)val;
    }
  }
}

extern "C" void kernel_launch(void* const* d_in, const int* in_sizes, int n_in,
                              void* d_out, int out_size, void* d_ws, size_t ws_size,
                              hipStream_t stream) {
  (void)in_sizes; (void)n_in; (void)out_size; (void)ws_size;
  const float* kp    = (const float*)d_in[0];
  const float* x     = (const float*)d_in[1];
  const float* W_pre = (const float*)d_in[2];
  const float* b_pre = (const float*)d_in[3];
  const float* um    = (const float*)d_in[4];
  const float* ut0   = (const float*)d_in[5];
  const float* ut1   = (const float*)d_in[6];
  const float* ut2   = (const float*)d_in[7];
  const float* ut3   = (const float*)d_in[8];
  const float* Wl3   = (const float*)d_in[9];
  const float* Wl2   = (const float*)d_in[10];
  const float* Wl1   = (const float*)d_in[11];
  const float* Wl0   = (const float*)d_in[12];
  const float* Wf    = (const float*)d_in[13];
  const float* bl3   = (const float*)d_in[14];
  const float* bl2   = (const float*)d_in[15];
  const float* bl1   = (const float*)d_in[16];
  const float* bl0   = (const float*)d_in[17];
  const float* bf    = (const float*)d_in[18];
  const int* idx0    = (const int*)d_in[19];
  const int* idx1    = (const int*)d_in[20];
  const int* idx2    = (const int*)d_in[21];
  const int* idx3    = (const int*)d_in[22];
  float* out = (float*)d_out;

  char* p = (char*)d_ws;
  auto carve = [&](size_t elems) {
    _Float16* r = (_Float16*)p;
    p += ((elems * sizeof(_Float16) + 255) & ~(size_t)255);
    return r;
  };
  _Float16* xs    = carve((size_t)1344 * 256);
  _Float16* x2    = carve((size_t)1344 * 256);
  _Float16* hA    = carve((size_t)64 * 195 * 256); // ping
  _Float16* hB    = carve((size_t)64 * 195 * 256); // pong
  _Float16* wpreh = carve((size_t)256 * 256);
  _Float16* wl3h  = carve((size_t)256 * 2304);
  _Float16* wl2h  = carve((size_t)128 * 2304);
  _Float16* wl1h  = carve((size_t)64 * 1152);
  _Float16* wl0h  = carve((size_t)32 * 576);
  _Float16* wfh   = carve((size_t)3 * 288);
  _Float16* umh   = carve((size_t)49 * 21);
  _Float16* ut0h  = carve((size_t)778 * 389);
  _Float16* ut1h  = carve((size_t)389 * 195);
  _Float16* ut2h  = carve((size_t)195 * 98);
  _Float16* ut3h  = carve((size_t)98 * 49);

  auto cvt = [&](_Float16* d, const float* s, int n) {
    k_f32_to_f16<<<dim3((n + 255) / 256), dim3(256), 0, stream>>>(d, s, n);
  };
  cvt(wpreh, W_pre, 256 * 256);
  cvt(wl3h, Wl3, 256 * 2304);
  cvt(wl2h, Wl2, 128 * 2304);
  cvt(wl1h, Wl1, 64 * 1152);
  cvt(wl0h, Wl0, 32 * 576);
  cvt(wfh, Wf, 3 * 288);
  cvt(umh, um, 49 * 21);
  cvt(ut0h, ut0, 778 * 389);
  cvt(ut1h, ut1, 389 * 195);
  cvt(ut2h, ut2, 195 * 98);
  cvt(ut3h, ut3, 98 * 49);

  // 1) bilinear sample of raw features (1x1 conv commuted past grid_sample)
  k_sample<<<dim3(1344), dim3(256), 0, stream>>>(kp, x, xs);
  // 2) x2 = xs @ W_pre^T + b_pre
  k_gemm<<<dim3(84, 4), dim3(TPB), 0, stream>>>(xs, wpreh, b_pre, x2, 1344, 256, 256);
  // 3) h49 = um @ x2
  k_upsample<<<dim3(64 * 4, 4), dim3(TPB), 0, stream>>>(umh, x2, hA, 49, 21, 256);
  // 4) layer 3
  k_upsample<<<dim3(64 * 7, 4), dim3(TPB), 0, stream>>>(ut3h, hA, hB, 98, 49, 256);
  k_dsconv<<<dim3(64 * 7, 4), dim3(TPB), 0, stream>>>(hB, idx3, wl3h, bl3, hA, nullptr,
                                                      98, 8, 256, 1);
  // 5) layer 2
  k_upsample<<<dim3(64 * 13, 4), dim3(TPB), 0, stream>>>(ut2h, hA, hB, 195, 98, 256);
  k_dsconv<<<dim3(64 * 13, 2), dim3(TPB), 0, stream>>>(hB, idx2, wl2h, bl2, hA, nullptr,
                                                       195, 8, 128, 1);
  // 6) layer 1
  k_upsample<<<dim3(64 * 25, 2), dim3(TPB), 0, stream>>>(ut1h, hA, hB, 389, 195, 128);
  k_dsconv<<<dim3(64 * 25, 1), dim3(TPB), 0, stream>>>(hB, idx1, wl1h, bl1, hA, nullptr,
                                                       389, 7, 64, 1);
  // 7) layer 0
  k_upsample<<<dim3(64 * 49, 1), dim3(TPB), 0, stream>>>(ut0h, hA, hB, 778, 389, 64);
  k_dsconv<<<dim3(64 * 49, 1), dim3(TPB), 0, stream>>>(hB, idx0, wl0h, bl0, hA, nullptr,
                                                       778, 6, 32, 1);
  // 8) head: spiral conv (32->3), f32 output
  k_dsconv<<<dim3(64 * 49, 1), dim3(TPB), 0, stream>>>(hA, idx0, wfh, bf, nullptr, out,
                                                       778, 5, 3, 0);
}
